// LossFun_42752104465123
// MI455X (gfx1250) — compile-verified, matrix operand order checked by reading, not run on hardware
//
#include <hip/hip_runtime.h>
#include <math.h>

typedef __attribute__((ext_vector_type(16))) _Float16     v16h;
typedef __attribute__((ext_vector_type(8)))  float        v8f;
typedef __attribute__((ext_vector_type(4)))  unsigned int v4u;
typedef __attribute__((ext_vector_type(8)))  int          v8i;
typedef __attribute__((ext_vector_type(4)))  int          v4i;

#define TPB     256
#define MAXOBJ  32
#define IOU_THR 0.5f

__device__ __forceinline__ float smooth_l1(float x) {
    float ax = fabsf(x);
    return ax < 1.0f ? 0.5f * x * x : ax - 0.5f;
}

// Fast reciprocal for IoU (feeds only a threshold compare and an argmax key).
__device__ __forceinline__ float fast_div(float a, float b) {
#if __has_builtin(__builtin_amdgcn_rcpf)
    return a * __builtin_amdgcn_rcpf(b);   // v_rcp_f32 + v_mul_f32 (TRANS co-exec)
#else
    return __fdividef(a, b);
#endif
}

__global__ void k_zero(float* acc) {
    if (threadIdx.x < 4) acc[threadIdx.x] = 0.0f;
}

// One block per batch: TDM-stage priors in LDS, match, encode,
// smooth-l1 (WMMA ones-reduction), logsumexp, per-prior lc.
__global__ __launch_bounds__(TPB)
void k_match(const float* __restrict__ loc_data,
             const float* __restrict__ conf_data,
             const float* __restrict__ targets,
             const float* __restrict__ priors,
             float* __restrict__ lc,
             int*   __restrict__ num_pos,
             float* __restrict__ acc,
             int N, int n_obj, int C)
{
    const int b   = blockIdx.x;
    const int tid = threadIdx.x;

    extern __shared__ float sPri[];          // N*4 floats: LDS copy of priors
    __shared__ float tx0[MAXOBJ], ty0[MAXOBJ], tx1[MAXOBJ], ty1[MAXOBJ];
    __shared__ float tarea[MAXOBJ], tlab[MAXOBJ];
    __shared__ unsigned long long sKey[MAXOBJ];
    __shared__ int   sBp[MAXOBJ];
    __shared__ float sWl[TPB / 32], sWc[TPB / 32];
    __shared__ int   sNp[TPB / 32];

    if (tid < n_obj) {
        const float* tp = targets + ((size_t)b * n_obj + tid) * 5;
        float x0 = tp[0], y0 = tp[1], x1 = tp[2], y1 = tp[3];
        tx0[tid] = x0; ty0[tid] = y0; tx1[tid] = x1; ty1[tid] = y1;
        tarea[tid] = (x1 - x0) * (y1 - y0);
        tlab[tid]  = tp[4];
        sKey[tid]  = 0ull;
    }

    // ---- Stage priors into LDS: one TDM DMA per block (wave 0 issues) ----
    const unsigned ne = (unsigned)(N * 4);   // f32 elements
#if __has_builtin(__builtin_amdgcn_tensor_load_to_lds)
    const bool use_tdm = (ne <= 0xFFFFu);    // tile_dim0 is 16-bit
    if (use_tdm) {
        if (tid < 32) {
            unsigned long long ga = (unsigned long long)(uintptr_t)priors;
            unsigned lds = (unsigned)(uintptr_t)sPri;
            // D# group0: count=1 | lds_addr | global_addr[56:0] | type=2
            v4u g0 = { 1u, lds,
                       (unsigned)(ga & 0xFFFFFFFFull),
                       ((unsigned)((ga >> 32) & 0x01FFFFFFull)) | (2u << 30) };
            // D# group1: data_size=4B; tensor_dim0=tile_dim0=ne; dim1=1; stride0=stride1=ne
            v8i g1 = { (int)(2u << 16),                      // data_size=2 (4 bytes)
                       (int)((ne & 0xFFFFu) << 16),          // tensor_dim0[15:0]
                       (int)((ne >> 16) | (1u << 16)),       // tensor_dim0[31:16] | tensor_dim1=1
                       (int)((ne & 0xFFFFu) << 16),          // tensor_dim1 hi=0 | tile_dim0
                       0,                                    // tile_dim1=0, tile_dim2=0 (unused)
                       (int)ne,                              // tensor_dim0_stride[31:0]
                       (int)((ne & 0xFFFFu) << 16),          // stride0 hi=0 | stride1[15:0]
                       (int)(ne >> 16) };                    // stride1[47:16]
            v4i g2 = { 0, 0, 0, 0 };
            v4i g3 = { 0, 0, 0, 0 };
#if defined(__clang_major__) && (__clang_major__ >= 23)
            v8i g4 = { 0, 0, 0, 0, 0, 0, 0, 0 };
            __builtin_amdgcn_tensor_load_to_lds(g0, g1, g2, g3, g4, 0);
#else
            __builtin_amdgcn_tensor_load_to_lds(g0, g1, g2, g3, 0);
#endif
#if __has_builtin(__builtin_amdgcn_s_wait_tensorcnt)
            __builtin_amdgcn_s_wait_tensorcnt(0);
#endif
        }
    } else
#endif
    {
        for (unsigned i = tid; i < ne; i += TPB) sPri[i] = priors[i];
    }
    __syncthreads();

    // ---- Phase A: per-truth argmax over priors (packed-key atomicMax) ----
    unsigned long long bk[16];
    #pragma unroll
    for (int t = 0; t < 16; ++t) bk[t] = 0ull;

    for (int i = tid; i < N; i += TPB) {
        const float* pp = sPri + (size_t)i * 4;
        float cx = pp[0], cy = pp[1], w = pp[2], h = pp[3];
        float px0 = cx - 0.5f * w, py0 = cy - 0.5f * h;
        float px1 = cx + 0.5f * w, py1 = cy + 0.5f * h;
        float parea = w * h;
        unsigned long long ik = (unsigned long long)(0xFFFFFFFFu - (unsigned)i);
        for (int t = 0; t < n_obj; ++t) {
            float ix = fminf(tx1[t], px1) - fmaxf(tx0[t], px0);
            float iy = fminf(ty1[t], py1) - fmaxf(ty0[t], py0);
            float inter = fmaxf(ix, 0.0f) * fmaxf(iy, 0.0f);
            float iou = fast_div(inter, tarea[t] + parea - inter);
            unsigned long long key = ((unsigned long long)__float_as_uint(iou) << 32) | ik;
            if (t < 16) { if (key > bk[t]) bk[t] = key; }
            else atomicMax(&sKey[t], key);
        }
    }
    #pragma unroll
    for (int t = 0; t < 16; ++t)
        if (t < n_obj) atomicMax(&sKey[t], bk[t]);
    __syncthreads();
    if (tid < n_obj)
        sBp[tid] = (int)(0xFFFFFFFFu - (unsigned)(sKey[tid] & 0xFFFFFFFFull));
    __syncthreads();

    // ---- Phase B: per-prior match + losses; WMMA ones-reduction ----
    union { v16h v; _Float16 h[16]; } ones, stg;
    #pragma unroll
    for (int j = 0; j < 16; ++j) { ones.h[j] = (_Float16)1.0f; stg.h[j] = (_Float16)0.0f; }
    v8f accl = {};
    float part16 = 0.0f;     // f32 accumulation over 16 iterations, flushed via slot 0
    int   cnt = 0;
    float cpos = 0.0f;
    int   npl  = 0;

    const int iters = (N + TPB - 1) / TPB;   // uniform trip count: EXEC all-1s at WMMA
    for (int it = 0; it < iters; ++it) {
        int i = it * TPB + tid;
        if (i < N) {
            const float* pp = sPri + (size_t)i * 4;
            float cx = pp[0], cy = pp[1], w = pp[2], h = pp[3];
            float px0 = cx - 0.5f * w, py0 = cy - 0.5f * h;
            float px1 = cx + 0.5f * w, py1 = cy + 0.5f * h;
            float parea = w * h;

            float bto = -1.0f; int bti = 0;
            for (int t = 0; t < n_obj; ++t) {
                float ix = fminf(tx1[t], px1) - fmaxf(tx0[t], px0);
                float iy = fminf(ty1[t], py1) - fmaxf(ty0[t], py0);
                float inter = fmaxf(ix, 0.0f) * fmaxf(iy, 0.0f);
                float iou = fast_div(inter, tarea[t] + parea - inter);
                if (iou > bto) { bto = iou; bti = t; }
            }
            // force-best-prior override (later truth index wins, like the fori_loop)
            for (int j = 0; j < n_obj; ++j)
                if (sBp[j] == i) { bto = 2.0f; bti = j; }

            int  cidx = (bto < IOU_THR) ? 0 : ((int)tlab[bti] + 1);
            bool pos  = cidx > 0;

            float mx0 = tx0[bti], my0 = ty0[bti], mx1 = tx1[bti], my1 = ty1[bti];
            float g0 = ((mx0 + mx1) * 0.5f - cx) / (0.1f * w);
            float g1 = ((my0 + my1) * 0.5f - cy) / (0.1f * h);
            float g2 = logf((mx1 - mx0) / w) / 0.2f;
            float g3 = logf((my1 - my0) / h) / 0.2f;
            const float* lp = loc_data + ((size_t)b * N + i) * 4;
            float sl = smooth_l1(lp[0] - g0) + smooth_l1(lp[1] - g1) +
                       smooth_l1(lp[2] - g2) + smooth_l1(lp[3] - g3);
            if (pos) { part16 += sl; npl++; }

            const float* cp = conf_data + ((size_t)b * N + i) * (size_t)C;
            __builtin_prefetch(cp + (size_t)TPB * C, 0, 1);  // global_prefetch_b8
            float m = cp[0];
            for (int k = 1; k < C; ++k) m = fmaxf(m, cp[k]);
            float s = 0.0f;
            for (int k = 0; k < C; ++k) s += __expf(cp[k] - m);
            float lse = m + __logf(s);
            float la  = lse - cp[cidx];
            lc[(size_t)b * N + i] = pos ? 0.0f : la;
            if (pos) cpos += la;
        }
        // uniform flush: constant-index slot 0 keeps staging in VGPRs (no LDS spill)
        if (++cnt == 16) {
            stg.h[0] = (_Float16)part16;
            accl = __builtin_amdgcn_wmma_f32_16x16x32_f16(
                false, stg.v, false, ones.v, (short)0, accl, false, false);
            part16 = 0.0f;
            cnt = 0;
        }
    }
    if (cnt != 0) {   // uniform tail flush
        stg.h[0] = (_Float16)part16;
        accl = __builtin_amdgcn_wmma_f32_16x16x32_f16(
            false, stg.v, false, ones.v, (short)0, accl, false, false);
    }
    // rows 0-7 live in lanes 0-15, rows 8-15 in lanes 16-31 (any column n works)
    float s8 = accl[0] + accl[1] + accl[2] + accl[3] + accl[4] + accl[5] + accl[6] + accl[7];
    float wtot = s8 + __shfl_xor(s8, 16, 32);

    for (int off = 16; off > 0; off >>= 1) {
        cpos += __shfl_down(cpos, off, 32);
        npl  += __shfl_down(npl,  off, 32);
    }
    int lane = tid & 31, wid = tid >> 5;
    if (lane == 0) { sWl[wid] = wtot; sWc[wid] = cpos; sNp[wid] = npl; }
    __syncthreads();
    if (tid == 0) {
        float bl = 0.0f, bc = 0.0f; int bn = 0;
        for (int w = 0; w < TPB / 32; ++w) { bl += sWl[w]; bc += sWc[w]; bn += sNp[w]; }
        atomicAdd(&acc[0], bl);
        atomicAdd(&acc[1], bc);
        num_pos[b] = bn;
    }
}

// One block per batch: exact 4-pass radix-select for the top-num_neg threshold,
// then sum of selected negatives (ties are sum-equivalent: T * thr).
__global__ __launch_bounds__(TPB)
void k_select(const float* __restrict__ lc,
              const int*   __restrict__ num_pos,
              float* __restrict__ acc,
              int N)
{
    const int b = blockIdx.x, tid = threadIdx.x;
    __shared__ unsigned hist[256];
    __shared__ unsigned sPrefix;
    __shared__ int sK;
    __shared__ float sW[TPB / 32];

    const float* row = lc + (size_t)b * N;
    int np = num_pos[b];
    int nn = min(3 * np, N - 1);
    if (nn <= 0) return;

    unsigned prefix = 0; int k = nn;
    for (int shift = 24; shift >= 0; shift -= 8) {
        hist[tid] = 0;
        __syncthreads();
        unsigned himask = (shift == 24) ? 0u : (0xFFFFFFFFu << (shift + 8));
        for (int i = tid; i < N; i += TPB) {
            unsigned u = __float_as_uint(row[i]);   // lc >= 0 -> bits order-preserving
            if ((u & himask) == (prefix & himask))
                atomicAdd(&hist[(u >> shift) & 255u], 1u);
        }
        __syncthreads();
        if (tid == 0) {
            int kk = k, bin = 255;
            for (; bin >= 0; --bin) {
                int c = (int)hist[bin];
                if (kk > c) kk -= c; else break;
            }
            if (bin < 0) bin = 0;
            sPrefix = prefix | ((unsigned)bin << shift);
            sK = kk;
        }
        __syncthreads();
        prefix = sPrefix; k = sK;
        __syncthreads();
    }

    float part = 0.0f;
    for (int i = tid; i < N; i += TPB) {
        float v = row[i];
        if (__float_as_uint(v) > prefix) part += v;
    }
    for (int off = 16; off > 0; off >>= 1) part += __shfl_down(part, off, 32);
    int lane = tid & 31, wid = tid >> 5;
    if (lane == 0) sW[wid] = part;
    __syncthreads();
    if (tid == 0) {
        float tot = 0.0f;
        for (int w = 0; w < TPB / 32; ++w) tot += sW[w];
        tot += (float)k * __uint_as_float(prefix);  // T tied elements at threshold
        atomicAdd(&acc[2], tot);
    }
}

__global__ void k_final(const int* __restrict__ num_pos,
                        const float* __restrict__ acc,
                        float* __restrict__ out, int B)
{
    __shared__ int sW[8];
    int tid = threadIdx.x;
    int s = 0;
    for (int i = tid; i < B; i += blockDim.x) s += num_pos[i];
    for (int off = 16; off > 0; off >>= 1) s += __shfl_down(s, off, 32);
    int lane = tid & 31, wid = tid >> 5;
    if (lane == 0) sW[wid] = s;
    __syncthreads();
    if (tid == 0) {
        int n = 0;
        for (int w = 0; w < (int)(blockDim.x / 32); ++w) n += sW[w];
        float fn = (float)max(n, 1);
        out[0] = acc[0] / fn;
        out[1] = (acc[1] + acc[2]) / fn;
    }
}

extern "C" void kernel_launch(void* const* d_in, const int* in_sizes, int n_in,
                              void* d_out, int out_size, void* d_ws, size_t ws_size,
                              hipStream_t stream)
{
    const float* loc  = (const float*)d_in[0];
    const float* conf = (const float*)d_in[1];
    const float* tgt  = (const float*)d_in[2];
    const float* pri  = (const float*)d_in[3];

    const int N     = in_sizes[3] / 4;             // priors: N x 4
    const int B     = in_sizes[0] / (N * 4);       // loc:    B x N x 4
    const int n_obj = in_sizes[2] / (B * 5);       // tgt:    B x n_obj x 5
    const int C     = in_sizes[1] / (B * N);       // conf:   B x N x C

    float* lc   = (float*)d_ws;                    // B*N floats (~4.5 MB, L2-resident)
    int*   npos = (int*)(lc + (size_t)B * N);      // B ints
    float* acc  = (float*)(npos + B);              // [loss_l, loss_c_pos, loss_c_neg, pad]
    float* out  = (float*)d_out;

    const size_t smem = (size_t)N * 4 * sizeof(float);  // LDS-resident priors (~140 KB)

    k_zero  <<<1, 32,  0,    stream>>>(acc);
    k_match <<<B, TPB, smem, stream>>>(loc, conf, tgt, pri, lc, npos, acc, N, n_obj, C);
    k_select<<<B, TPB, 0,    stream>>>(lc, npos, acc, N);
    k_final <<<1, 128, 0,    stream>>>(npos, acc, out, B);
}
